// IterativeARLayer_91250875171387
// MI455X (gfx1250) — compile-verified, matrix operand order checked by reading, not run on hardware
//
#include <hip/hip_runtime.h>
#include <math.h>

// ---------------------------------------------------------------------------
// IterativeARLayer.decode on MI455X (gfx1250).
// The per-site "scan" has no loop-carried dependence on x; every step reads
// only z. So all 511 steps run in parallel as masked GEMMs
//   (1024 x (s+1)) @ ((s+1) x 128)
// using v_wmma_f32_16x16x32_bf16 (bf16 in, f32 accumulate).
//
// Tiling: block = (step s, 64-row batch tile); 8 waves each own a 16-wide
// slice of H=128 and carry FOUR 16x16 accumulators, so each fetched B chunk
// (16 strided b32 loads, the expensive part) feeds 4 WMMAs. A-tile loads are
// coalesced b128. K loop = unmasked full 32-chunks + one value-masked tail.
// ---------------------------------------------------------------------------

typedef __attribute__((ext_vector_type(16))) __bf16 v16bf;
typedef __attribute__((ext_vector_type(8)))  float  v8f;

#define B_SZ 1024
#define D_SZ 512
#define H_SZ 128
#define S_SZ 511
#define PI_F 3.14159265358979323846f

#define M_TILE 64   // batch rows per block
#define N_ACC  4    // 16-row accumulators per wave

// -------------------------------------------------------------------- init --
__global__ void ar_init_kernel(const float* __restrict__ z,
                               float* __restrict__ x,
                               float* __restrict__ logdet) {
    int t = blockIdx.x * blockDim.x + threadIdx.x;
    if (t < B_SZ) {
        x[(size_t)t * D_SZ] = z[(size_t)t * D_SZ];  // site 0 passthrough
        logdet[t] = 0.0f;                           // for atomic fallback
    }
}

// Load one lane's 16 A elements (two contiguous runs of 8 floats) as b128s,
// optionally value-mask the autoregressive tail, convert to bf16 lane layout.
template <bool MASKED>
__device__ __forceinline__ v16bf ar_load_a(const float* __restrict__ zrow,
                                           int k0, int klo, int rem) {
    float af[16];
    const float4* p0 = (const float4*)(zrow + k0 + klo);
    const float4* p1 = (const float4*)(zrow + k0 + 16 + klo);
    ((float4*)af)[0] = p0[0];
    ((float4*)af)[1] = p0[1];
    ((float4*)af)[2] = p1[0];
    ((float4*)af)[3] = p1[1];
    if (MASKED) {  // selects on values only -> v_cndmask, no exec branches
        #pragma unroll
        for (int j = 0; j < 8; ++j) {
            af[j]     = ((klo + j)      < rem) ? af[j]     : 0.0f;
            af[8 + j] = ((16 + klo + j) < rem) ? af[8 + j] : 0.0f;
        }
    }
    v16bf av;
    #pragma unroll
    for (int i = 0; i < 16; ++i) av[i] = (__bf16)af[i];
    return av;
}

// One K-chunk of 32: shared B tile + 4 M-tiles -> 4 WMMAs.
template <bool MASKED>
__device__ __forceinline__ void ar_chunk(const float* __restrict__ zrow,
                                         const float* __restrict__ wcol,
                                         int k0, int klo, int rem,
                                         v8f acc[N_ACC]) {
    // ---- B tile: strided fp32 column reads of W1[s] (stride H) -----------
    float bf[16];
    #pragma unroll
    for (int j = 0; j < 8; ++j) {
        bf[j]     = wcol[(size_t)(k0 + klo + j) * H_SZ];
        bf[8 + j] = wcol[(size_t)(k0 + 16 + klo + j) * H_SZ];
    }
    v16bf bv;
    #pragma unroll
    for (int i = 0; i < 16; ++i) bv[i] = (__bf16)bf[i];

    #pragma unroll
    for (int a = 0; a < N_ACC; ++a) {
        v16bf av = ar_load_a<MASKED>(zrow + (size_t)(16 * a) * D_SZ,
                                     k0, klo, rem);
        acc[a] = __builtin_amdgcn_wmma_f32_16x16x32_bf16(
                     false, av, false, bv, (short)0, acc[a], false, false);
    }
}

// -------------------------------------------------------------- main step --
__launch_bounds__(256)
__global__ void ar_step_kernel(const float* __restrict__ z,
                               const float* __restrict__ W1,
                               const float* __restrict__ b1,
                               const float* __restrict__ W2,
                               const float* __restrict__ b2,
                               float* __restrict__ x,
                               float* __restrict__ logdet,
                               float* __restrict__ ld_ws,
                               int deterministic) {
    const int s      = blockIdx.x;            // step; transforms site s+1
    const int m_base = blockIdx.y * M_TILE;   // batch-row tile
    const int tid    = threadIdx.x;
    const int wave   = tid >> 5;
    const int lane   = tid & 31;
    const int lhalf  = lane >> 4;             // 0: lanes 0-15, 1: lanes 16-31
    const int l15    = lane & 15;
    const int n      = wave * 16 + l15;       // this lane's H column
    const int klo    = lhalf * 8;             // ISA 7.12.2 16-bit lane K offset

    __shared__ float s_pa[8][M_TILE];
    __shared__ float s_pb[8][M_TILE];

    const float* __restrict__ zrow = z + (size_t)(m_base + l15) * D_SZ;
    const float* __restrict__ wcol = W1 + (size_t)s * D_SZ * H_SZ + n;

    v8f acc[N_ACC] = {};
    const int total = s + 1;              // live K extent
    const int nfull = total >> 5;         // unmasked 32-chunks
    const int rem   = total & 31;         // masked tail length

    for (int kc = 0; kc < nfull; ++kc) {
        const int k0 = kc << 5;
        __builtin_prefetch(wcol + (size_t)(k0 + 32 + klo) * H_SZ, 0, 1);
        ar_chunk<false>(zrow, wcol, k0, klo, 0, acc);
    }
    if (rem) {
        ar_chunk<true>(zrow, wcol, nfull << 5, klo, rem, acc);
    }

    // h = relu(acc + b1); fold W2 into per-lane alpha/beta partials.
    const float b1v = b1[(size_t)s * H_SZ + n];
    const float w2a = W2[((size_t)s * H_SZ + n) * 2 + 0];
    const float w2b = W2[((size_t)s * H_SZ + n) * 2 + 1];

    float pa[N_ACC][8], pb[N_ACC][8];
    #pragma unroll
    for (int a = 0; a < N_ACC; ++a) {
        #pragma unroll
        for (int r = 0; r < 8; ++r) {
            float h = acc[a][r] + b1v;
            h = h > 0.0f ? h : 0.0f;
            pa[a][r] = h * w2a;
            pb[a][r] = h * w2b;
        }
    }

    // C layout: vgpr r / lanes 0-15 -> row r, lanes 16-31 -> row r+8.
    // xor masks 1,2,4,8 reduce over the 16 N-lanes within each half-wave.
    #pragma unroll
    for (int off = 1; off < 16; off <<= 1) {
        #pragma unroll
        for (int a = 0; a < N_ACC; ++a) {
            #pragma unroll
            for (int r = 0; r < 8; ++r) {
                pa[a][r] += __shfl_xor(pa[a][r], off, 32);
                pb[a][r] += __shfl_xor(pb[a][r], off, 32);
            }
        }
    }
    if (l15 == 0) {  // lane 0 -> rows r, lane 16 -> rows r+8 (plain stores)
        #pragma unroll
        for (int a = 0; a < N_ACC; ++a) {
            #pragma unroll
            for (int r = 0; r < 8; ++r) {
                s_pa[wave][16 * a + lhalf * 8 + r] = pa[a][r];
                s_pb[wave][16 * a + lhalf * 8 + r] = pb[a][r];
            }
        }
    }
    __syncthreads();

    // NCP diffeomorphism on site idx = s+1 for the 64 rows of this tile.
    if (tid < M_TILE) {
        float alpha = b2[s * 2 + 0];
        float beta  = b2[s * 2 + 1];
        #pragma unroll
        for (int w = 0; w < 8; ++w) {     // sum the 8 wave partials
            alpha += s_pa[w][tid];
            beta  += s_pb[w][tid];
        }
        const int   idx  = s + 1;
        const int   row  = m_base + tid;
        const float a    = expf(alpha);
        const float phi  = z[(size_t)row * D_SZ + idx];
        const float u    = tanf(0.5f * (phi - PI_F));
        const float v    = fmaf(a, u, beta);
        const float outv = 2.0f * atanf(v) + PI_F;
        const float ld   = alpha + log1pf(u * u) - log1pf(v * v);

        x[(size_t)row * D_SZ + idx] = outv;
        if (deterministic) {
            ld_ws[(size_t)s * B_SZ + row] = ld;   // run-stable reduction
        } else {
            atomicAdd(&logdet[row], ld);          // global_atomic_add_f32
        }
    }
}

// ------------------------------------------------------- log_det reduction --
__global__ void ar_logdet_reduce_kernel(const float* __restrict__ ld_ws,
                                        float* __restrict__ logdet) {
    int row = blockIdx.x * blockDim.x + threadIdx.x;
    if (row < B_SZ) {
        float acc = 0.0f;
        for (int sstep = 0; sstep < S_SZ; ++sstep)   // coalesced across rows
            acc += ld_ws[(size_t)sstep * B_SZ + row];
        logdet[row] = acc;
    }
}

// ------------------------------------------------------------------ launch --
extern "C" void kernel_launch(void* const* d_in, const int* in_sizes, int n_in,
                              void* d_out, int out_size, void* d_ws, size_t ws_size,
                              hipStream_t stream) {
    const float* z  = (const float*)d_in[0];
    const float* W1 = (const float*)d_in[1];
    const float* b1 = (const float*)d_in[2];
    const float* W2 = (const float*)d_in[3];
    const float* b2 = (const float*)d_in[4];

    float* x      = (float*)d_out;                    // (B, D) flat
    float* logdet = x + (size_t)B_SZ * D_SZ;          // (B,)
    float* ld_ws  = (float*)d_ws;

    const size_t need = (size_t)S_SZ * B_SZ * sizeof(float);
    const int deterministic = (ws_size >= need) ? 1 : 0;

    ar_init_kernel<<<(B_SZ + 255) / 256, 256, 0, stream>>>(z, x, logdet);

    dim3 grid(S_SZ, B_SZ / M_TILE);     // 511 steps x 16 batch tiles
    ar_step_kernel<<<grid, 256, 0, stream>>>(z, W1, b1, W2, b2,
                                             x, logdet, ld_ws, deterministic);

    if (deterministic) {
        ar_logdet_reduce_kernel<<<(B_SZ + 255) / 256, 256, 0, stream>>>(ld_ws, logdet);
    }
}
